// MultiHeadSelfAttention_80315888435698
// MI455X (gfx1250) — matrix operand order repaired and checked by reading, not compiled, not run on hardware
//
#include <hip/hip_runtime.h>
#include <hip/hip_bf16.h>
#include <math.h>

typedef __attribute__((ext_vector_type(8)))  _Float16 v8h;
typedef __attribute__((ext_vector_type(16))) _Float16 v16h;
typedef __attribute__((ext_vector_type(8)))  float    v8f;

#define EMBED 512
#define HEADS 8
#define HDIM  64
#define SEQ   2048
#define BATCH 8
#define TOK   (BATCH * SEQ)

// ---------------------------------------------------------------------------
// CDNA5 primitives
// ---------------------------------------------------------------------------
__device__ __forceinline__ v8f wmma_f16(v16h a, v16h b, v8f c) {
  return __builtin_amdgcn_wmma_f32_16x16x32_f16(
      /*neg_a=*/false, a, /*neg_b=*/false, b,
      /*c_mod=*/(short)0, c, /*reuse_a=*/false, /*reuse_b=*/false);
}

// Async global->LDS 16-byte copy (per lane), tracked by ASYNCcnt.
// lds_addr: low 32 bits of a generic pointer to __shared__ (== LDS offset).
__device__ __forceinline__ void async_b128(unsigned int lds_addr,
                                           const void* gaddr) {
  asm volatile("global_load_async_to_lds_b128 %0, %1, off"
               :: "v"(lds_addr), "v"(gaddr)
               : "memory");
}
__device__ __forceinline__ void wait_async() {
  asm volatile("s_wait_asynccnt 0x0" ::: "memory");
}
__device__ __forceinline__ unsigned int lds_off(const void* p) {
  return (unsigned int)(unsigned long long)p;
}

__device__ __forceinline__ v8f v8f_zero() {
  v8f z;
#pragma unroll
  for (int i = 0; i < 8; ++i) z[i] = 0.0f;
  return z;
}

// A-fragment (16x32 f16, MxK), source row-major [M][K], `stride` halves/row.
// ISA: lanes 0-15 -> M=lane, K={0..7,16..23}; lanes 16-31 -> M=lane-16,
// K={8..15,24..31}. Two contiguous 16-B loads per lane.
__device__ __forceinline__ v16h load_afrag(const _Float16* base, int stride) {
  const int lane = threadIdx.x & 31;
  const int r    = lane & 15;
  const int kb   = (lane >> 4) << 3;
  const _Float16* p = base + r * stride + kb;
  v8h lo = *reinterpret_cast<const v8h*>(p);
  v8h hi = *reinterpret_cast<const v8h*>(p + 16);
  return __builtin_shufflevector(lo, hi, 0, 1, 2, 3, 4, 5, 6, 7,
                                 8, 9, 10, 11, 12, 13, 14, 15);
}

// B-fragment (32x16 f16, KxN) supplied as Bt row-major [N][K].
// ISA: lanes 0-15 -> N=lane, K={0..7,16..23};
// lanes 16-31 -> N=lane-16, K={8..15,24..31}.
__device__ __forceinline__ v16h load_bfrag(const _Float16* base, int stride) {
  const int lane = threadIdx.x & 31;
  const int n    = lane & 15;
  const int k0   = (lane >> 4) << 3;
  const _Float16* p = base + n * stride + k0;
  v8h lo = *reinterpret_cast<const v8h*>(p);
  v8h hi = *reinterpret_cast<const v8h*>(p + 16);
  return __builtin_shufflevector(lo, hi, 0, 1, 2, 3, 4, 5, 6, 7,
                                 8, 9, 10, 11, 12, 13, 14, 15);
}

// Convert 16 contiguous fp32 -> 16 f16 into LDS.
__device__ __forceinline__ void cvt16_f32_to_f16(const float4* s4, _Float16* dst) {
  float4 f0 = s4[0], f1 = s4[1], f2 = s4[2], f3 = s4[3];
  dst[0]  = (_Float16)f0.x; dst[1]  = (_Float16)f0.y;
  dst[2]  = (_Float16)f0.z; dst[3]  = (_Float16)f0.w;
  dst[4]  = (_Float16)f1.x; dst[5]  = (_Float16)f1.y;
  dst[6]  = (_Float16)f1.z; dst[7]  = (_Float16)f1.w;
  dst[8]  = (_Float16)f2.x; dst[9]  = (_Float16)f2.y;
  dst[10] = (_Float16)f2.z; dst[11] = (_Float16)f2.w;
  dst[12] = (_Float16)f3.x; dst[13] = (_Float16)f3.y;
  dst[14] = (_Float16)f3.z; dst[15] = (_Float16)f3.w;
}

// ---------------------------------------------------------------------------
// Kernel 1: fused QKV projection. C[m][n] = X[m][:] . W[n][:]  (x @ W^T)
// blockIdx.z selects (Q,Wq)->qh, (K,Wk)->kh, (V,Wv)->vt (transposed).
// q pre-scaled by 1/tau. KC=64 -> 16 WMMA per barrier pair.
// ---------------------------------------------------------------------------
__global__ __launch_bounds__(256) void qkv_proj_kernel(
    const float* __restrict__ Qx, const float* __restrict__ Kx,
    const float* __restrict__ Vx, const float* __restrict__ Wq,
    const float* __restrict__ Wk, const float* __restrict__ Wv,
    const float* __restrict__ tau,
    _Float16* __restrict__ qh, _Float16* __restrict__ kh,
    _Float16* __restrict__ vt) {
  __shared__ __align__(16) _Float16 As[128 * 64];
  __shared__ __align__(16) _Float16 Bs[128 * 64];

  const int which = blockIdx.z;
  const float* X = (which == 0) ? Qx : (which == 1) ? Kx : Vx;
  const float* W = (which == 0) ? Wq : (which == 1) ? Wk : Wv;

  const int m0 = blockIdx.x * 128;
  const int n0 = blockIdx.y * 128;
  const int tid = threadIdx.x;
  const int wv = tid >> 5, lane = tid & 31;
  const int wm = wv & 3, wn = wv >> 2;  // wave tile: 32(M) x 64(N)
  const int lr = tid >> 1;              // 0..127
  const int lc = (tid & 1) << 5;        // 0 or 32 (halves)

  v8f acc[2][4];
#pragma unroll
  for (int mi = 0; mi < 2; ++mi)
#pragma unroll
    for (int ni = 0; ni < 4; ++ni) acc[mi][ni] = v8f_zero();

  for (int k0 = 0; k0 < EMBED; k0 += 64) {
    __syncthreads();
    {
      const float4* s = reinterpret_cast<const float4*>(
          X + (size_t)(m0 + lr) * EMBED + k0 + lc);
      cvt16_f32_to_f16(s, &As[lr * 64 + lc]);
      cvt16_f32_to_f16(s + 4, &As[lr * 64 + lc + 16]);
    }
    {
      const float4* s = reinterpret_cast<const float4*>(
          W + (size_t)(n0 + lr) * EMBED + k0 + lc);
      cvt16_f32_to_f16(s, &Bs[lr * 64 + lc]);
      cvt16_f32_to_f16(s + 4, &Bs[lr * 64 + lc + 16]);
    }
    __syncthreads();

    v16h af[2][2], bf[4][2];
#pragma unroll
    for (int mi = 0; mi < 2; ++mi)
#pragma unroll
      for (int kk = 0; kk < 2; ++kk)
        af[mi][kk] = load_afrag(&As[(wm * 32 + mi * 16) * 64 + kk * 32], 64);
#pragma unroll
    for (int ni = 0; ni < 4; ++ni)
#pragma unroll
      for (int kk = 0; kk < 2; ++kk)
        bf[ni][kk] = load_bfrag(&Bs[(wn * 64 + ni * 16) * 64 + kk * 32], 64);
#pragma unroll
    for (int mi = 0; mi < 2; ++mi)
#pragma unroll
      for (int ni = 0; ni < 4; ++ni) {
        acc[mi][ni] = wmma_f16(af[mi][0], bf[ni][0], acc[mi][ni]);
        acc[mi][ni] = wmma_f16(af[mi][1], bf[ni][1], acc[mi][ni]);
      }
  }

  const float itau = (which == 0) ? (1.0f / tau[0]) : 1.0f;
  const int half = lane >> 4, lcol = lane & 15;
#pragma unroll
  for (int mi = 0; mi < 2; ++mi) {
#pragma unroll
    for (int ni = 0; ni < 4; ++ni) {
      const int gmBase = m0 + wm * 32 + mi * 16 + 8 * half;
      const int gn = n0 + wn * 64 + ni * 16 + lcol;
      const int h = gn >> 6, d = gn & 63;
#pragma unroll
      for (int i = 0; i < 8; ++i) {
        const int gm = gmBase + i;
        const int b = gm >> 11, ns = gm & 2047;
        const size_t head = (size_t)(b * HEADS + h);
        const _Float16 hv = (_Float16)(acc[mi][ni][i] * itau);
        if (which == 0)
          qh[(head * SEQ + ns) * HDIM + d] = hv;
        else if (which == 1)
          kh[(head * SEQ + ns) * HDIM + d] = hv;
        else
          vt[(head * HDIM + d) * SEQ + ns] = hv;  // V stored transposed
      }
    }
  }
}

// ---------------------------------------------------------------------------
// Kernel 2: flash-style attention per (b,h).
// 128 query rows/block (8 waves x 16). Keys streamed in chunks of 128 with
// DOUBLE-BUFFERED async global->LDS staging (ASYNCcnt) overlapping the WMMAs.
// ---------------------------------------------------------------------------
__global__ __launch_bounds__(256) void attn_kernel(
    const _Float16* __restrict__ qh, const _Float16* __restrict__ kh,
    const _Float16* __restrict__ vt, const int* __restrict__ mask,
    _Float16* __restrict__ oat) {
  __shared__ __align__(16) _Float16 Ks[2][128 * 64];   // [key][d]
  __shared__ __align__(16) _Float16 Vs[2][64 * 128];   // [d][key]
  __shared__ __align__(16) _Float16 Ps[8 * 16 * 128];  // per-wave [row][key]

  const int bh = blockIdx.y;
  const int b = bh >> 3, h = bh & 7;
  const int q0 = blockIdx.x * 128;
  const int tid = threadIdx.x;
  const int wv = tid >> 5, lane = tid & 31;
  const int half = lane >> 4, lcol = lane & 15;

  // Stage one 128-key chunk (K and V^T) via async LDS loads.
  auto stage = [&](int sb, int kc) {
    {
      const int key = tid >> 1, c = (tid & 1) * 32;
      const _Float16* g = kh + ((size_t)bh * SEQ + kc + key) * HDIM + c;
      const unsigned int l = lds_off(&Ks[sb][key * 64 + c]);
      async_b128(l, g);
      async_b128(l + 16, g + 8);
      async_b128(l + 32, g + 16);
      async_b128(l + 48, g + 24);
    }
    {
      const int dd = tid >> 2, c = (tid & 3) * 32;
      const _Float16* g = vt + ((size_t)bh * HDIM + dd) * SEQ + kc + c;
      const unsigned int l = lds_off(&Vs[sb][dd * 128 + c]);
      async_b128(l, g);
      async_b128(l + 16, g + 8);
      async_b128(l + 32, g + 16);
      async_b128(l + 48, g + 24);
    }
  };

  // q fragments for this wave's 16 rows (constant across key loop)
  const _Float16* qbase = qh + ((size_t)bh * SEQ + q0 + wv * 16) * HDIM;
  const v16h aq0 = load_afrag(qbase, HDIM);       // k-dims 0..31
  const v16h aq1 = load_afrag(qbase + 32, HDIM);  // k-dims 32..63

  v8f o[4];
#pragma unroll
  for (int t = 0; t < 4; ++t) o[t] = v8f_zero();
  float rmax[8], rsum[8];
#pragma unroll
  for (int i = 0; i < 8; ++i) { rmax[i] = -INFINITY; rsum[i] = 0.0f; }

  const int* mrow = mask + (size_t)b * SEQ * SEQ;  // [N][N], head-broadcast
  _Float16* Pw = &Ps[wv * 16 * 128];

  stage(0, 0);
  int buf = 0;
  for (int kc = 0; kc < SEQ; kc += 128) {
    wait_async();     // our async chunk is in LDS
    __syncthreads();  // everyone's chunk is in LDS; prev reads of buf^1 done
    if (kc + 128 < SEQ) stage(buf ^ 1, kc + 128);  // overlap next chunk

    const _Float16* Kc = &Ks[buf][0];
    const _Float16* Vc = &Vs[buf][0];

    // ---- scores: 16 x 128 per wave (q pre-scaled by 1/tau) ----
    float s[8][8];  // [key-subtile j][row i]
#pragma unroll
    for (int j = 0; j < 8; ++j) {
      v16h b0 = load_bfrag(&Kc[(j * 16) * 64], 64);
      v16h b1 = load_bfrag(&Kc[(j * 16) * 64 + 32], 64);
      v8f sc = v8f_zero();
      sc = wmma_f16(aq0, b0, sc);
      sc = wmma_f16(aq1, b1, sc);
#pragma unroll
      for (int i = 0; i < 8; ++i) s[j][i] = sc[i];
    }

    // ---- mask + chunk row-max ----
    float cmax[8];
#pragma unroll
    for (int i = 0; i < 8; ++i) cmax[i] = -INFINITY;
#pragma unroll
    for (int j = 0; j < 8; ++j) {
#pragma unroll
      for (int i = 0; i < 8; ++i) {
        const int qrow = q0 + wv * 16 + i + 8 * half;
        const int key = kc + j * 16 + lcol;
        if (mrow[(size_t)qrow * SEQ + key] == 0) s[j][i] = -INFINITY;
        cmax[i] = fmaxf(cmax[i], s[j][i]);
      }
    }
#pragma unroll
    for (int i = 0; i < 8; ++i)
#pragma unroll
      for (int m = 1; m < 16; m <<= 1)
        cmax[i] = fmaxf(cmax[i], __shfl_xor(cmax[i], m, 32));

    // ---- online softmax update ----
#pragma unroll
    for (int i = 0; i < 8; ++i) {
      const float nm = fmaxf(rmax[i], cmax[i]);
      const float scale = __expf(rmax[i] - nm);
      rmax[i] = nm;
      float ls = 0.0f;
#pragma unroll
      for (int j = 0; j < 8; ++j) {
        const float p = __expf(s[j][i] - nm);
        s[j][i] = p;
        ls += p;
      }
#pragma unroll
      for (int m = 1; m < 16; m <<= 1) ls += __shfl_xor(ls, m, 32);
      rsum[i] = rsum[i] * scale + ls;
#pragma unroll
      for (int t = 0; t < 4; ++t) o[t][i] *= scale;
    }

    // ---- relayout P (accumulator layout -> A layout) via per-wave LDS ----
#pragma unroll
    for (int j = 0; j < 8; ++j)
#pragma unroll
      for (int i = 0; i < 8; ++i)
        Pw[(i + 8 * half) * 128 + j * 16 + lcol] = (_Float16)s[j][i];
    __syncthreads();  // (per-wave region; conservative ordering)

    // ---- O += P @ V ----
#pragma unroll
    for (int k0 = 0; k0 < 128; k0 += 32) {
      v16h pf = load_afrag(&Pw[k0], 128);
#pragma unroll
      for (int t = 0; t < 4; ++t) {
        v16h vf = load_bfrag(&Vc[(t * 16) * 128 + k0], 128);
        o[t] = wmma_f16(pf, vf, o[t]);
      }
    }
    buf ^= 1;
  }

  // ---- normalize; write [B,N,EMBED] f16 (A-side of output projection) ----
#pragma unroll
  for (int t = 0; t < 4; ++t) {
#pragma unroll
    for (int i = 0; i < 8; ++i) {
      const int qrow = q0 + wv * 16 + i + 8 * half;
      const int d = t * 16 + lcol;
      const float val = o[t][i] / rsum[i];
      oat[((size_t)b * SEQ + qrow) * EMBED + h * HDIM + d] = (_Float16)val;
    }
  }
}

// ---------------------------------------------------------------------------
// Kernel 3: output projection  Y = O @ Wo^T + bo   (fp32 out).
// A tile (already f16) staged via async global->LDS; B converted via VGPRs.
// ---------------------------------------------------------------------------
__global__ __launch_bounds__(256) void out_proj_kernel(
    const _Float16* __restrict__ oat, const float* __restrict__ Wo,
    const float* __restrict__ bo, float* __restrict__ out) {
  __shared__ __align__(16) _Float16 As[128 * 64];
  __shared__ __align__(16) _Float16 Bs[128 * 64];

  const int m0 = blockIdx.x * 128;
  const int n0 = blockIdx.y * 128;
  const int tid = threadIdx.x;
  const int wv = tid >> 5, lane = tid & 31;
  const int wm = wv & 3, wn = wv >> 2;
  const int lr = tid >> 1;
  const int lc = (tid & 1) << 5;

  v8f acc[2][4];
#pragma unroll
  for (int mi = 0; mi < 2; ++mi)
#pragma unroll
    for (int ni = 0; ni < 4; ++ni) acc[mi][ni] = v8f_zero();

  for (int k0 = 0; k0 < EMBED; k0 += 64) {
    __syncthreads();
    {  // A tile: f16, async straight into LDS
      const _Float16* g = oat + (size_t)(m0 + lr) * EMBED + k0 + lc;
      const unsigned int l = lds_off(&As[lr * 64 + lc]);
      async_b128(l, g);
      async_b128(l + 16, g + 8);
      async_b128(l + 32, g + 16);
      async_b128(l + 48, g + 24);
    }
    {  // B tile: fp32 -> f16 via VGPRs
      const float4* s = reinterpret_cast<const float4*>(
          Wo + (size_t)(n0 + lr) * EMBED + k0 + lc);
      cvt16_f32_to_f16(s, &Bs[lr * 64 + lc]);
      cvt16_f32_to_f16(s + 4, &Bs[lr * 64 + lc + 16]);
    }
    wait_async();
    __syncthreads();

    v16h af[2][2], bf[4][2];
#pragma unroll
    for (int mi = 0; mi < 2; ++mi)
#pragma unroll
      for (int kk = 0; kk < 2; ++kk)
        af[mi][kk] = load_afrag(&As[(wm * 32 + mi * 16) * 64 + kk * 32], 64);
#pragma unroll
    for (int ni = 0; ni < 4; ++ni)
#pragma unroll
      for (int kk = 0; kk < 2; ++kk)
        bf[ni][kk] = load_bfrag(&Bs[(wn * 64 + ni * 16) * 64 + kk * 32], 64);
#pragma unroll
    for (int mi = 0; mi < 2; ++mi)
#pragma unroll
      for (int ni = 0; ni < 4; ++ni) {
        acc[mi][ni] = wmma_f16(af[mi][0], bf[ni][0], acc[mi][ni]);
        acc[mi][ni] = wmma_f16(af[mi][1], bf[ni][1], acc[mi][ni]);
      }
  }

  const int half = lane >> 4, lcol = lane & 15;
#pragma unroll
  for (int mi = 0; mi < 2; ++mi) {
#pragma unroll
    for (int ni = 0; ni < 4; ++ni) {
      const int gmBase = m0 + wm * 32 + mi * 16 + 8 * half;
      const int gn = n0 + wn * 64 + ni * 16 + lcol;
      const float bias = bo[gn];
#pragma unroll
      for (int i = 0; i < 8; ++i)
        out[(size_t)(gmBase + i) * EMBED + gn] = acc[mi][ni][i] + bias;
    }
  }
}

// ---------------------------------------------------------------------------
extern "C" void kernel_launch(void* const* d_in, const int* in_sizes, int n_in,
                              void* d_out, int out_size, void* d_ws,
                              size_t ws_size, hipStream_t stream) {
  (void)in_sizes; (void)n_in; (void)out_size; (void)ws_size;
  const float* Q    = (const float*)d_in[0];
  const float* K    = (const float*)d_in[1];
  const float* V    = (const float*)d_in[2];
  const int*   mask = (const int*)d_in[3];
  const float* Wq   = (const float*)d_in[4];
  const float* Wk   = (const float*)d_in[5];
  const float* Wv   = (const float*)d_in[6];
  const float* Wo   = (const float*)d_in[7];
  const float* bo   = (const float*)d_in[8];
  const float* tau  = (const float*)d_in[9];

  char* ws = (char*)d_ws;
  const size_t per = (size_t)BATCH * HEADS * SEQ * HDIM * sizeof(_Float16);
  _Float16* qh  = (_Float16*)(ws);            // [B,H,N,D]  q / tau
  _Float16* kh  = (_Float16*)(ws + per);      // [B,H,N,D]
  _Float16* vt  = (_Float16*)(ws + 2 * per);  // [B,H,D,N]  (transposed)
  _Float16* oat = (_Float16*)(ws + 3 * per);  // [B,N,E]    attention output

  dim3 blk(256);
  qkv_proj_kernel<<<dim3(TOK / 128, EMBED / 128, 3), blk, 0, stream>>>(
      Q, K, V, Wq, Wk, Wv, tau, qh, kh, vt);
  attn_kernel<<<dim3(SEQ / 128, BATCH * HEADS), blk, 0, stream>>>(
      qh, kh, vt, mask, oat);
  out_proj_kernel<<<dim3(TOK / 128, EMBED / 128), blk, 0, stream>>>(
      oat, Wo, bo, (float*)d_out);
}
